// GraphEncoder_89945205113533
// MI455X (gfx1250) — compile-verified, matrix operand order checked by reading, not run on hardware
//
#include <hip/hip_runtime.h>
#include <hip/hip_bf16.h>
#include <math.h>

// ---------------------------------------------------------------------------
// GraphEncoder (2-layer GAT + LN + gated pooling) for MI455X (gfx1250).
//
// softmax(l_i + r_j) over masked j == exp(r_j)·mask / Σ, so each GAT layer is
// ONE 0/1-masked GEMM:
//     V[j,0:128] = exp(r_j - m) * h_j ;  V[j,128] = exp(r_j - m)
//     [num|den]  = adj_f16 @ V_f16     (v_wmma_f32_16x16x32_f16, fp32 accum)
//     out        = elu(num / den)
// adj (256MB int32, non-temporal) is the only big HBM stream (~11us/layer).
// k_attn processes 32 rows/wave so each B fragment feeds 2 WMMAs,
// halving L2-side Vt traffic vs 16 rows/wave.
// ---------------------------------------------------------------------------

typedef __attribute__((ext_vector_type(16))) _Float16 v16h;
typedef __attribute__((ext_vector_type(8)))  float    v8f;
typedef __attribute__((ext_vector_type(4)))  int      v4i;

#define NN 8192
#define DD 128
#define NT 144          // 128 value cols + 1 denom col + 15 zero pad = 9 tiles
#define KSPLIT 4
#define KC (NN / KSPLIT)

union Frag16 { v16h h; unsigned int u[8]; uint4 q[2]; };

__device__ __forceinline__ unsigned int pack01(int a, int b) {
  // {a,b} in {0,1}  ->  packed f16 pair {a?1.0:0.0, b?1.0:0.0}
  return (a ? 0x3C00u : 0u) | (b ? 0x3C000000u : 0u);
}
__device__ __forceinline__ unsigned int packff(float a, float b) {
  union { _Float16 h[2]; unsigned int u; } t;
  t.h[0] = (_Float16)a; t.h[1] = (_Float16)b;
  return t.u;
}

// ---------------- W transpose + f16 convert: Wt[c][k] = W[k][c] -------------
__global__ __launch_bounds__(256) void k_transW(const float* __restrict__ W,
                                                _Float16* __restrict__ Wt) {
  int idx = blockIdx.x * 256 + threadIdx.x;   // 16384 total
  int c = idx >> 7, k = idx & 127;
  Wt[(size_t)c * DD + k] = (_Float16)W[(size_t)k * DD + c];
}

// ---------------- H = X @ W  via WMMA (one wave per 16-row block) -----------
__global__ __launch_bounds__(32) void k_gemmW(const float* __restrict__ X,
                                              const _Float16* __restrict__ Wt,
                                              float* __restrict__ H) {
  const int lane = threadIdx.x;
  const int hs   = lane >> 4;          // half-wave select (ISA 16-bit A layout)
  const int mr   = lane & 15;
  const int row  = blockIdx.x * 16 + mr;
  const v8f zero = {0.f,0.f,0.f,0.f,0.f,0.f,0.f,0.f};
  v8f acc[8];
#pragma unroll
  for (int t = 0; t < 8; ++t) acc[t] = zero;

  for (int k0 = 0; k0 < DD; k0 += 32) {
    // A fragment: rows of X, K groups {k0+hs*8 ..+7} and {+16}
    const float* xr = X + (size_t)row * DD + k0 + hs * 8;
    float4 f0 = *(const float4*)(xr + 0);
    float4 f1 = *(const float4*)(xr + 4);
    float4 f2 = *(const float4*)(xr + 16);
    float4 f3 = *(const float4*)(xr + 20);
    Frag16 a;
    a.u[0]=packff(f0.x,f0.y); a.u[1]=packff(f0.z,f0.w);
    a.u[2]=packff(f1.x,f1.y); a.u[3]=packff(f1.z,f1.w);
    a.u[4]=packff(f2.x,f2.y); a.u[5]=packff(f2.z,f2.w);
    a.u[6]=packff(f3.x,f3.y); a.u[7]=packff(f3.z,f3.w);
#pragma unroll
    for (int t = 0; t < 8; ++t) {
      const _Float16* vr = Wt + (size_t)(16*t + mr) * DD + k0 + hs * 8;
      Frag16 b;
      b.q[0] = *(const uint4*)(vr);
      b.q[1] = *(const uint4*)(vr + 16);
      acc[t] = __builtin_amdgcn_wmma_f32_16x16x32_f16(
          false, a.h, false, b.h, (short)0, acc[t], false, false);
    }
  }
#pragma unroll
  for (int t = 0; t < 8; ++t)
#pragma unroll
    for (int g = 0; g < 8; ++g)
      H[(size_t)(blockIdx.x*16 + g + hs*8) * DD + 16*t + mr] = acc[t][g];
}

// ---------------- r[row] = h[row,:] . a_hi  (one warp per row) --------------
__global__ __launch_bounds__(256) void k_rowdot(const float* __restrict__ H,
                                                const float* __restrict__ ahi,
                                                float* __restrict__ r) {
  int w = blockIdx.x * 8 + (threadIdx.x >> 5);
  int lane = threadIdx.x & 31;
  const float* hr = H + (size_t)w * DD;
  float s = 0.f;
#pragma unroll
  for (int j = 0; j < 4; ++j) s += hr[lane + 32*j] * ahi[lane + 32*j];
#pragma unroll
  for (int m = 16; m > 0; m >>= 1) s += __shfl_xor(s, m, 32);
  if (lane == 0) r[w] = s;
}

// ---------------- global max of r --------------------------------------------
__global__ __launch_bounds__(256) void k_maxr(const float* __restrict__ r,
                                              float* __restrict__ misc) {
  __shared__ float sd[256];
  int t = threadIdx.x;
  float m = -3.4e38f;
  for (int i = t; i < NN; i += 256) m = fmaxf(m, r[i]);
  sd[t] = m; __syncthreads();
  for (int k = 128; k > 0; k >>= 1) {
    if (t < k) sd[t] = fmaxf(sd[t], sd[t + k]);
    __syncthreads();
  }
  if (t == 0) misc[0] = sd[0];
}

// ---------------- Vt[c][k] build (transposed f16, pad cols zero) ------------
__global__ __launch_bounds__(256) void k_buildV(const float* __restrict__ H,
                                                const float* __restrict__ r,
                                                const float* __restrict__ misc,
                                                _Float16* __restrict__ Vt) {
  int k = blockIdx.x * 256 + threadIdx.x;
  int c = blockIdx.y;                 // 0..143
  float e = expf(r[k] - misc[0]);     // in (0,1]
  float val;
  if      (c < DD)  val = e * H[(size_t)k * DD + c];
  else if (c == DD) val = e;
  else              val = 0.f;
  Vt[(size_t)c * NN + k] = (_Float16)val;
}

// ---------------- big GEMM: partials[ks] = adj_f16 @ Vt ---------------------
// one wave per (32-row block, K-split); 2 x 9 accumulator tiles (144 cols)
// each B fragment load feeds 2 WMMAs -> halves L2 Vt traffic
__global__ __launch_bounds__(32) void k_attn(const int* __restrict__ adj,
                                             const _Float16* __restrict__ Vt,
                                             float* __restrict__ P) {
  const int lane = threadIdx.x;
  const int hs   = lane >> 4;
  const int mr   = lane & 15;
  const int rb   = blockIdx.x;        // 32-row block
  const int ks   = blockIdx.y;
  const int row0 = rb * 32 + mr;
  const int* arow0 = adj + (size_t)row0 * NN;
  const int* arow1 = arow0 + (size_t)16 * NN;
  const v8f zero = {0.f,0.f,0.f,0.f,0.f,0.f,0.f,0.f};
  v8f acc0[9], acc1[9];
#pragma unroll
  for (int t = 0; t < 9; ++t) { acc0[t] = zero; acc1[t] = zero; }

  const int kbeg = ks * KC, kend = kbeg + KC;
  for (int k0 = kbeg; k0 < kend; k0 += 32) {
    const int base = k0 + hs * 8;
    // adj is a 256MB single-use stream: non-temporal so Vt stays L2-resident
    v4i i0 = __builtin_nontemporal_load((const v4i*)(arow0 + base + 0));
    v4i i1 = __builtin_nontemporal_load((const v4i*)(arow0 + base + 4));
    v4i i2 = __builtin_nontemporal_load((const v4i*)(arow0 + base + 16));
    v4i i3 = __builtin_nontemporal_load((const v4i*)(arow0 + base + 20));
    v4i j0 = __builtin_nontemporal_load((const v4i*)(arow1 + base + 0));
    v4i j1 = __builtin_nontemporal_load((const v4i*)(arow1 + base + 4));
    v4i j2 = __builtin_nontemporal_load((const v4i*)(arow1 + base + 16));
    v4i j3 = __builtin_nontemporal_load((const v4i*)(arow1 + base + 20));
    Frag16 a0, a1;
    a0.u[0]=pack01(i0[0],i0[1]); a0.u[1]=pack01(i0[2],i0[3]);
    a0.u[2]=pack01(i1[0],i1[1]); a0.u[3]=pack01(i1[2],i1[3]);
    a0.u[4]=pack01(i2[0],i2[1]); a0.u[5]=pack01(i2[2],i2[3]);
    a0.u[6]=pack01(i3[0],i3[1]); a0.u[7]=pack01(i3[2],i3[3]);
    a1.u[0]=pack01(j0[0],j0[1]); a1.u[1]=pack01(j0[2],j0[3]);
    a1.u[2]=pack01(j1[0],j1[1]); a1.u[3]=pack01(j1[2],j1[3]);
    a1.u[4]=pack01(j2[0],j2[1]); a1.u[5]=pack01(j2[2],j2[3]);
    a1.u[6]=pack01(j3[0],j3[1]); a1.u[7]=pack01(j3[2],j3[3]);
#pragma unroll
    for (int t = 0; t < 9; ++t) {
      const _Float16* vr = Vt + (size_t)(16*t + mr) * NN + base;
      Frag16 b;
      b.q[0] = *(const uint4*)(vr);
      b.q[1] = *(const uint4*)(vr + 16);
      acc0[t] = __builtin_amdgcn_wmma_f32_16x16x32_f16(
          false, a0.h, false, b.h, (short)0, acc0[t], false, false);
      acc1[t] = __builtin_amdgcn_wmma_f32_16x16x32_f16(
          false, a1.h, false, b.h, (short)0, acc1[t], false, false);
    }
  }
  float* Pb = P + ((size_t)ks * NN + (size_t)rb * 32) * NT;
#pragma unroll
  for (int t = 0; t < 9; ++t)
#pragma unroll
    for (int g = 0; g < 8; ++g) {
      Pb[(size_t)(g + hs*8)      * NT + 16*t + mr] = acc0[t][g];
      Pb[(size_t)(g + hs*8 + 16) * NT + 16*t + mr] = acc1[t][g];
    }
}

// ---------------- combine K-splits, divide, elu -----------------------------
__global__ __launch_bounds__(128) void k_combine(const float* __restrict__ P,
                                                 float* __restrict__ Hout) {
  int row = blockIdx.x, c = threadIdx.x;
  float num = 0.f, den = 0.f;
#pragma unroll
  for (int ks = 0; ks < KSPLIT; ++ks) {
    const float* pr = P + ((size_t)ks * NN + row) * NT;
    num += pr[c];
    den += pr[DD];
  }
  float v = num / den;
  Hout[(size_t)row * DD + c] = (v > 0.f) ? v : expm1f(v);
}

// ---------------- LayerNorm + gate scores -----------------------------------
__global__ __launch_bounds__(128) void k_layernorm(const float* __restrict__ H,
                                                   const float* __restrict__ g,
                                                   const float* __restrict__ b,
                                                   const float* __restrict__ gw,
                                                   const float* __restrict__ gb,
                                                   float* __restrict__ HN,
                                                   float* __restrict__ scores) {
  __shared__ float sd[128];
  int row = blockIdx.x, t = threadIdx.x;
  float v = H[(size_t)row * DD + t];
  sd[t] = v; __syncthreads();
  for (int s = 64; s > 0; s >>= 1) { if (t < s) sd[t] += sd[t+s]; __syncthreads(); }
  float mu = sd[0] * (1.f / DD); __syncthreads();
  float d = v - mu;
  sd[t] = d * d; __syncthreads();
  for (int s = 64; s > 0; s >>= 1) { if (t < s) sd[t] += sd[t+s]; __syncthreads(); }
  float var = sd[0] * (1.f / DD); __syncthreads();
  float hn = d * rsqrtf(var + 1e-5f) * g[t] + b[t];
  HN[(size_t)row * DD + t] = hn;
  sd[t] = hn * gw[t]; __syncthreads();
  for (int s = 64; s > 0; s >>= 1) { if (t < s) sd[t] += sd[t+s]; __syncthreads(); }
  if (t == 0) scores[row] = sd[0] + gb[0];
}

// ---------------- node-score softmax stats (max, Z) -------------------------
__global__ __launch_bounds__(256) void k_scorestats(const float* __restrict__ s,
                                                    float* __restrict__ misc) {
  __shared__ float sd[256];
  int t = threadIdx.x;
  float mv = -3.4e38f;
  for (int i = t; i < NN; i += 256) mv = fmaxf(mv, s[i]);
  sd[t] = mv; __syncthreads();
  for (int k = 128; k > 0; k >>= 1) { if (t < k) sd[t] = fmaxf(sd[t], sd[t+k]); __syncthreads(); }
  float smax = sd[0]; __syncthreads();
  float z = 0.f;
  for (int i = t; i < NN; i += 256) z += expf(s[i] - smax);
  sd[t] = z; __syncthreads();
  for (int k = 128; k > 0; k >>= 1) { if (t < k) sd[t] += sd[t+k]; __syncthreads(); }
  if (t == 0) { misc[1] = smax; misc[2] = sd[0]; }
}

// ---------------- node weights -> d_out[128..] ------------------------------
__global__ __launch_bounds__(256) void k_weights(const float* __restrict__ s,
                                                 const float* __restrict__ misc,
                                                 float* __restrict__ outw) {
  int i = blockIdx.x * 256 + threadIdx.x;
  outw[i] = expf(s[i] - misc[1]) / misc[2];
}

// ---------------- graph embedding: emb[c] = sum_row w[row]*HN[row,c] --------
__global__ __launch_bounds__(256) void k_emb(const float* __restrict__ HN,
                                             const float* __restrict__ w,
                                             float* __restrict__ out) {
  __shared__ float sd[256];
  int c = blockIdx.x, t = threadIdx.x;
  float s = 0.f;
  for (int row = t; row < NN; row += 256) s += w[row] * HN[(size_t)row * DD + c];
  sd[t] = s; __syncthreads();
  for (int k = 128; k > 0; k >>= 1) { if (t < k) sd[t] += sd[t+k]; __syncthreads(); }
  if (t == 0) out[c] = sd[0];
}

// ---------------------------------------------------------------------------
// workspace layout (bytes)
static constexpr size_t OFF_H0   = 0;
static constexpr size_t OFF_H1   = OFF_H0 + (size_t)NN * DD * 4;       // 4MB
static constexpr size_t OFF_R    = OFF_H1 + (size_t)NN * DD * 4;       // 8MB
static constexpr size_t OFF_S    = OFF_R + (size_t)NN * 4;
static constexpr size_t OFF_MISC = OFF_S + (size_t)NN * 4;
static constexpr size_t OFF_WT   = OFF_MISC + 256;
static constexpr size_t OFF_VT   = OFF_WT + (size_t)DD * DD * 2;
static constexpr size_t OFF_P    = OFF_VT + (size_t)NT * NN * 2;
// total = OFF_P + KSPLIT*NN*NT*4  (~28.3 MiB)

static void run_layer(const float* X, const float* W, const float* a,
                      const int* adj, char* ws, float* Htmp, float* Hout,
                      hipStream_t stream) {
  float*     rbuf = (float*)(ws + OFF_R);
  float*     misc = (float*)(ws + OFF_MISC);
  _Float16*  Wt   = (_Float16*)(ws + OFF_WT);
  _Float16*  Vt   = (_Float16*)(ws + OFF_VT);
  float*     P    = (float*)(ws + OFF_P);

  k_transW <<<64, 256, 0, stream>>>(W, Wt);
  k_gemmW  <<<NN/16, 32, 0, stream>>>(X, Wt, Htmp);
  k_rowdot <<<NN/8, 256, 0, stream>>>(Htmp, a + DD, rbuf);
  k_maxr   <<<1, 256, 0, stream>>>(rbuf, misc);
  k_buildV <<<dim3(NN/256, NT), 256, 0, stream>>>(Htmp, rbuf, misc, Vt);
  k_attn   <<<dim3(NN/32, KSPLIT), 32, 0, stream>>>(adj, Vt, P);
  k_combine<<<NN, 128, 0, stream>>>(P, Hout);
}

extern "C" void kernel_launch(void* const* d_in, const int* in_sizes, int n_in,
                              void* d_out, int out_size, void* d_ws, size_t ws_size,
                              hipStream_t stream) {
  (void)in_sizes; (void)n_in; (void)out_size; (void)ws_size;
  const float* x    = (const float*)d_in[0];
  const int*   adj  = (const int*)d_in[1];
  const float* W1   = (const float*)d_in[2];
  const float* a1   = (const float*)d_in[3];
  const float* W2   = (const float*)d_in[4];
  const float* a2   = (const float*)d_in[5];
  const float* ln_g = (const float*)d_in[6];
  const float* ln_b = (const float*)d_in[7];
  const float* gw   = (const float*)d_in[8];
  const float* gb   = (const float*)d_in[9];
  float* out = (float*)d_out;                    // [0:128]=emb, [128:8320]=weights

  char*  ws   = (char*)d_ws;
  float* H0   = (float*)(ws + OFF_H0);
  float* H1   = (float*)(ws + OFF_H1);
  float* scr  = (float*)(ws + OFF_S);
  float* misc = (float*)(ws + OFF_MISC);

  // layer 1: x -> H1   (scratch H0)
  run_layer(x,  W1, a1, adj, ws, H0, H1, stream);
  // layer 2: H1 -> H1  (scratch H0; H1 input consumed by k_gemmW first)
  run_layer(H1, W2, a2, adj, ws, H0, H1, stream);

  // LayerNorm + gating (HN stored in H0)
  k_layernorm <<<NN, 128, 0, stream>>>(H1, ln_g, ln_b, gw, gb, H0, scr);
  k_scorestats<<<1, 256, 0, stream>>>(scr, misc);
  k_weights   <<<NN/256, 256, 0, stream>>>(scr, misc, out + DD);
  k_emb       <<<DD, 256, 0, stream>>>(H0, out + DD, out);
}
